// SpikingNeuronLayer_34076270526892
// MI455X (gfx1250) — matrix operand order repaired and checked by reading, not compiled, not run on hardware
//
#include <hip/hip_runtime.h>

// ---------------------------------------------------------------------------
// Fused SNN layer for MI455X (gfx1250):
//   out[n,b] = sum_t [ mp_t(n,b) >= 1 ],  mp_t = 0.9*mp_{t-1} + (W @ x[:,t,:]^T)
// GEMM runs on bf16 WMMA with hi/lo split-2 fp32 emulation (3 wmma per K=32).
// W tile (64 rows) lives in LDS as bf16 hi/lo planes for all 512 timesteps.
// x tiles stream through LDS via the Tensor Data Mover (double buffered),
// then are converted fp32 -> bf16 hi/lo ONCE per chunk by the whole
// workgroup (instead of per-wave), keeping the inner loop XDL-bound.
// ---------------------------------------------------------------------------

#define T_STEPS 512u
#define I_DIM   1024u
#define N_DIM   1024u
#define B_DIM   128u
#define DECAY   0.9f
#define THRESH  1.0f

typedef __bf16 bf16_t;
typedef bf16_t        v16bf __attribute__((ext_vector_type(16)));
typedef float         v8f   __attribute__((ext_vector_type(8)));
typedef float         v4f   __attribute__((ext_vector_type(4)));
typedef unsigned int  v4u   __attribute__((ext_vector_type(4)));
typedef int           v4i   __attribute__((ext_vector_type(4)));
typedef int           v8i   __attribute__((ext_vector_type(8)));

// LDS layout (dynamic shared, 304 KB total):
//   [0,128K)          Whi  : 64 x 1024 bf16
//   [128K,256K)       Wlo  : 64 x 1024 bf16
//   [256K,256K+32K)   xraw : 2 x (32 b-rows x 128 k fp32)  TDM double buffer
//   [288K,288K+16K)   xbf  : hi plane + lo plane, 32 x 128 bf16 each
#define WHI_OFF    0u
#define WLO_OFF    (64u * 1024u * 2u)
#define XRAW_OFF   (2u * 64u * 1024u * 2u)
#define XRAW_BYTES 16384u
#define XBF_OFF    (XRAW_OFF + 2u * XRAW_BYTES)
#define XBFLO_OFF  (XBF_OFF + 8192u)
#define LDS_TOTAL  (XBF_OFF + 2u * 8192u)

union ABFrag { v16bf v; v4u u[2]; };

// Issue one TDM 2-D tile load: x[b0..b0+31, t, c*128 .. c*128+127] -> LDS.
// D# per CDNA5 ISA §8: group0 = {count|flags, lds_addr, gaddr_lo, gaddr_hi|type=2}
// group1 packs data_size=4B, tensor/tile dims, row stride = T*I elements.
__device__ __forceinline__ void tdm_load_x(const float* xbase, unsigned b0,
                                           unsigned t, unsigned c,
                                           unsigned lds_byte_addr) {
  unsigned long long ga = (unsigned long long)(const void*)xbase +
      (((((unsigned long long)b0) * T_STEPS + t) * I_DIM +
        (unsigned long long)c * 128ull) << 2);
  v4u g0;
  g0[0] = 1u;                                   // count=1 valid descriptor
  g0[1] = lds_byte_addr;                        // lds_addr
  g0[2] = (unsigned)ga;                         // global_addr[31:0]
  g0[3] = (unsigned)(ga >> 32) | (2u << 30);    // global_addr[56:32] | type=2
  v8i g1;
  g1[0] = (int)(2u << 16);                      // data_size = 2 (4 bytes)
  g1[1] = (int)(128u << 16);                    // tensor_dim0 = 128
  g1[2] = (int)(32u << 16);                     // tensor_dim1 = 32
  g1[3] = (int)(128u << 16);                    // tile_dim0 = 128
  g1[4] = (int)32u;                             // tile_dim1 = 32, tile_dim2 = 0
  g1[5] = (int)(T_STEPS * I_DIM);               // tensor_dim0_stride = 524288
  g1[6] = 0;                                    // stride0 hi / stride1 lo
  g1[7] = 0;
  v4i z4; z4[0] = z4[1] = z4[2] = z4[3] = 0;    // groups 2/3 unused (2-D)
  v8i z8;
#pragma unroll
  for (int j = 0; j < 8; ++j) z8[j] = 0;
  __builtin_amdgcn_tensor_load_to_lds(g0, g1, z4, z4, z8, 0);
}

// Convert 4 fp32 lane values into bf16 hi/lo fragment slots [base, base+3].
__device__ __forceinline__ void cvt4(v4f f, ABFrag& hiF, ABFrag& loF,
                                     unsigned base) {
#pragma unroll
  for (unsigned j = 0; j < 4; ++j) {
    bf16_t h = (bf16_t)f[j];
    hiF.v[base + j] = h;
    loF.v[base + j] = (bf16_t)(f[j] - (float)h);
  }
}

__global__ __launch_bounds__(256, 1)
void snn_fused_kernel(const float* __restrict__ x, const float* __restrict__ w,
                      float* __restrict__ out) {
  extern __shared__ char smem[];
  bf16_t* whi   = (bf16_t*)(smem + WHI_OFF);
  bf16_t* wlo   = (bf16_t*)(smem + WLO_OFF);
  bf16_t* xbfhi = (bf16_t*)(smem + XBF_OFF);
  bf16_t* xbflo = (bf16_t*)(smem + XBFLO_OFF);

  const unsigned tid  = threadIdx.x;
  const unsigned lane = tid & 31u;
  const unsigned wave = tid >> 5;
  const unsigned n0 = blockIdx.x * 64u;   // 16 tiles over N
  const unsigned b0 = blockIdx.y * 32u;   // 4 tiles over B
  const unsigned wt = wave & 3u;          // n sub-tile (4)
  const unsigned wb = wave >> 2;          // b sub-tile (2)
  // A-matrix layout (16x32 bf16): lane&15 = row m; lanes<16 K{0-7,16-23},
  // lanes>=16 K{8-15,24-31}.
  const unsigned mrow  = wt * 16u + (lane & 15u);
  const unsigned koffA = (lane & 16u) ? 8u : 0u;
  // B-matrix layout (32x16 bf16): col = lane&15; lanes<16 K{0-7,16-23},
  // lanes>=16 K{8-15,24-31}.
  const unsigned koffB = (lane & 16u) ? 8u : 0u;
  const unsigned brow  = wb * 16u + (lane & 15u);

  const unsigned smemBase = (unsigned)(size_t)(void*)smem;

  // Kick off the first TDM load before (and overlapped with) W conversion.
  if (wave == 0) tdm_load_x(x, b0, 0u, 0u, smemBase + XRAW_OFF);

  // Convert this WG's 64xI weight slice to bf16 hi/lo planes in LDS (once).
  for (unsigned idx = tid; idx < 64u * I_DIM; idx += 256u) {
    const unsigned r = idx >> 10, i = idx & 1023u;
    const float wv = w[(size_t)(n0 + r) * I_DIM + i];
    const bf16_t h = (bf16_t)wv;
    whi[idx] = h;
    wlo[idx] = (bf16_t)(wv - (float)h);
  }

  v8f mp = {};   // membrane potential, WMMA C/D layout
  v8f cnt = {};  // spike counts
  unsigned s = 0;
  for (unsigned t = 0; t < T_STEPS; ++t) {
    v8f cur = {};
    for (unsigned c = 0; c < 8u; ++c, ++s) {
      if (wave == 0) __builtin_amdgcn_s_wait_tensorcnt(0);
      __syncthreads();  // xraw[s&1] ready; everyone past chunk s-1 compute
      if (wave == 0 && (s + 1u) < T_STEPS * 8u) {
        const unsigned sn = s + 1u;
        tdm_load_x(x, b0, sn >> 3, sn & 7u,
                   smemBase + XRAW_OFF + (sn & 1u) * XRAW_BYTES);
      }
      // Cooperative fp32 -> bf16 hi/lo conversion: 4096 elements, 16/thread.
      {
        const float* xraw = (const float*)(smem + XRAW_OFF +
                                           (s & 1u) * XRAW_BYTES);
        const unsigned e = tid * 16u;
        ABFrag hiF, loF;
        cvt4(*(const v4f*)(xraw + e +  0u), hiF, loF,  0u);
        cvt4(*(const v4f*)(xraw + e +  4u), hiF, loF,  4u);
        cvt4(*(const v4f*)(xraw + e +  8u), hiF, loF,  8u);
        cvt4(*(const v4f*)(xraw + e + 12u), hiF, loF, 12u);
        v4u* dh = (v4u*)(xbfhi + e);
        v4u* dl = (v4u*)(xbflo + e);
        dh[0] = hiF.u[0]; dh[1] = hiF.u[1];
        dl[0] = loF.u[0]; dl[1] = loF.u[1];
      }
      __syncthreads();  // bf16 planes ready

      const bf16_t* arow_hi = whi + (size_t)mrow * I_DIM + c * 128u + koffA;
      const bf16_t* arow_lo = wlo + (size_t)mrow * I_DIM + c * 128u + koffA;
      const bf16_t* brow_hi = xbfhi + brow * 128u + koffB;
      const bf16_t* brow_lo = xbflo + brow * 128u + koffB;
#pragma unroll
      for (unsigned kk = 0; kk < 4u; ++kk) {
        const unsigned k0 = kk * 32u;
        ABFrag ah, al, bh, bl;
        ah.u[0] = *(const v4u*)(arow_hi + k0);
        ah.u[1] = *(const v4u*)(arow_hi + k0 + 16u);
        al.u[0] = *(const v4u*)(arow_lo + k0);
        al.u[1] = *(const v4u*)(arow_lo + k0 + 16u);
        bh.u[0] = *(const v4u*)(brow_hi + k0);
        bh.u[1] = *(const v4u*)(brow_hi + k0 + 16u);
        bl.u[0] = *(const v4u*)(brow_lo + k0);
        bl.u[1] = *(const v4u*)(brow_lo + k0 + 16u);
        // fp32-emulating split: Whi*Xhi + Whi*Xlo + Wlo*Xhi
        cur = __builtin_amdgcn_wmma_f32_16x16x32_bf16(
            false, ah.v, false, bh.v, (short)0, cur, false, false);
        cur = __builtin_amdgcn_wmma_f32_16x16x32_bf16(
            false, ah.v, false, bl.v, (short)0, cur, false, false);
        cur = __builtin_amdgcn_wmma_f32_16x16x32_bf16(
            false, al.v, false, bh.v, (short)0, cur, false, false);
      }
    }
    // Leaky integrate-and-fire update, fused on the accumulator registers.
#pragma unroll
    for (unsigned r = 0; r < 8u; ++r) {
      float m = DECAY * mp[r] + cur[r];
      const float sp = (m >= THRESH) ? 1.0f : 0.0f;
      cnt[r] += sp;
      mp[r] = m * (1.0f - sp);
    }
  }

  // C/D layout: VGPR r holds row M = r (lanes 0-15) or r+8 (lanes 16-31).
#pragma unroll
  for (unsigned r = 0; r < 8u; ++r) {
    const unsigned n = n0 + wt * 16u + r + ((lane & 16u) ? 8u : 0u);
    const unsigned b = b0 + wb * 16u + (lane & 15u);
    out[(size_t)n * B_DIM + b] = cnt[r];
  }
}

extern "C" void kernel_launch(void* const* d_in, const int* in_sizes, int n_in,
                              void* d_out, int out_size, void* d_ws,
                              size_t ws_size, hipStream_t stream) {
  (void)in_sizes; (void)n_in; (void)d_ws; (void)ws_size; (void)out_size;
  const float* x = (const float*)d_in[0];   // [128, 512, 1024]
  const float* w = (const float*)d_in[1];   // [1024, 1024]
  float* out = (float*)d_out;               // [1024, 128]
  dim3 grid(N_DIM / 64u, B_DIM / 32u, 1);   // 16 x 4 workgroups
  dim3 block(256, 1, 1);                    // 8 waves
  snn_fused_kernel<<<grid, block, LDS_TOTAL, stream>>>(x, w, out);
}
